// GraphFormation_71244917506580
// MI455X (gfx1250) — compile-verified, matrix operand order checked by reading, not run on hardware
//
#include <hip/hip_runtime.h>

typedef __attribute__((ext_vector_type(16))) _Float16 v16h;
typedef __attribute__((ext_vector_type(8)))  _Float16 v8h;
typedef __attribute__((ext_vector_type(8)))  float    v8f;

#define DIM 64
#define NNODE 100

#define WMMA_F16(A,B,C) __builtin_amdgcn_wmma_f32_16x16x32_f16(false,(A),false,(B),(short)0,(C),false,false)

__device__ __forceinline__ float clamp1(float x){ return fminf(1.f, fmaxf(-1.f, x)); }

// Sparse local Cox-de Boor on the uniform grid grid[i] = -1.5 + 0.125*i.
// u = (x+1.5)*8 in [4,20]; only basis indices (si-4..si) are non-zero.
// Writes the 32-slot fp16 feature strip for one row into LDS:
//   slot0 = neg, slot1 = pos, slot 2+kk = basis_kk, slots 22..31 = 0.
__device__ __forceinline__ void stage_feats(float x, _Float16* srow){
  float xc  = clamp1(x);
  float pos = fmaxf(xc, 0.f);
  float neg = xc - pos;
  float u   = (xc + 1.5f) * 8.f;
  float sf  = floorf(u);
  float t   = u - sf;
  int   si  = (int)sf;                     // 4..20
  float N[5]; N[0] = 1.f;
#pragma unroll
  for (int jj = 1; jj <= 4; ++jj){
    float inv = 1.f / (float)jj;
    float saved = 0.f;
#pragma unroll
    for (int r = 0; r < jj; ++r){
      float temp = N[r] * inv;
      N[r] = saved + ((float)(r + 1) - t) * temp;     // right[r+1] = (r+1)-t
      saved = (t + (float)(jj - 1 - r)) * temp;       // left[jj-r] = t+(jj-r-1)
    }
    N[jj] = saved;
  }
  v8h z = {};
  *(v8h*)(srow)      = z;                  // zero slots 0..7
  *(v8h*)(srow + 8)  = z;                  // zero slots 8..15
  *(v8h*)(srow + 16) = z;                  // zero slots 16..23
  srow[0] = (_Float16)neg;
  srow[1] = (_Float16)pos;
  int base = si - 4;
#pragma unroll
  for (int r = 0; r < 5; ++r){
    int slot = base + r;
    if (slot < 20) srow[2 + slot] = (_Float16)N[r];   // 5 dynamic ds_store_b16
  }
}

// A fragment per the CDNA5 16-bit A layout:
// lane<16: elements 0..7 -> K 0..7, 8..15 -> K 16..23 ; lane>=16: K += 8.
__device__ __forceinline__ v16h load_afrag(const _Float16* srow, bool hi){
  int o = hi ? 8 : 0;
  v8h a0 = *(const v8h*)(srow + o);
  v8h a1 = *(const v8h*)(srow + 16 + o);
  return __builtin_shufflevector(a0, a1, 0,1,2,3,4,5,6,7,8,9,10,11,12,13,14,15);
}

// ---------------------------------------------------------------- feat (max-pool of |enc|)
__global__ void feat_kernel(const float* __restrict__ x, float* __restrict__ feat){
  int t = blockIdx.x * blockDim.x + threadIdx.x;          // 2*512*100
  if (t >= 2 * 512 * 100) return;
  int n = t % 100, c = (t / 100) % 512, b = t / (100 * 512);
  const float* p = x + ((size_t)(b * 512 + c)) * 1000 + n * 10;
  float mx = fabsf(p[0]);
#pragma unroll
  for (int w = 1; w < 10; ++w) mx = fmaxf(mx, fabsf(p[w]));
  feat[(b * 100 + n) * 512 + c] = mx;
}

// ---------------------------------------------------------------- projection + PE
__global__ __launch_bounds__(64) void proj_kernel(const float* __restrict__ feat,
    const float* __restrict__ W, const float* __restrict__ bias,
    const float* __restrict__ pe, float* __restrict__ out){
  __shared__ float sf[512];
  int bn = blockIdx.x;            // 0..199 = b*100+n
  int n  = bn % 100;
  int o  = threadIdx.x;
  for (int c = o; c < 512; c += 64) sf[c] = feat[bn * 512 + c];
  __syncthreads();
  float acc = 0.f;
  for (int c = 0; c < 512; ++c) acc += sf[c] * W[o * 512 + c];
  out[bn * DIM + o] = acc + bias[o] + pe[n * DIM + o];
}

// ---------------------------------------------------------------- weight prep (fp16, B-fragment order)
// wmat layout: [j=64][otile=4][lane=32][e=16] halves
__global__ void prep_w2_kernel(const float* __restrict__ prelu,
    const float* __restrict__ basew, const float* __restrict__ splinew,
    const float* __restrict__ coeffs, int out_f, _Float16* __restrict__ wmat){
  int t = blockIdx.x * blockDim.x + threadIdx.x;
  if (t >= 64 * 4 * 32) return;
  int lane = t & 31, ot = (t >> 5) & 3, j = t >> 7;
  int ncol = lane & 15; bool hi = lane >= 16;
  int o = ot * 16 + ncol;
  v16h v;
#pragma unroll
  for (int e = 0; e < 16; ++e){
    int k = (hi ? 8 : 0) + (e & 7) + ((e >= 8) ? 16 : 0);
    float w = 0.f;
    if (o < out_f){
      int oj = o * DIM + j;
      if      (k == 0) w = basew[oj] * prelu[oj];
      else if (k == 1) w = basew[oj];
      else if (k < 22) w = splinew[oj] * coeffs[oj * 20 + (k - 2)];
    }
    v[e] = (_Float16)w;
  }
  *(v16h*)(wmat + ((size_t)(j * 4 + ot) * 32 + lane) * 16) = v;
}

__global__ void wmean_kernel(const float* __restrict__ watt, float* __restrict__ wm){
  int d = threadIdx.x;
  if (d < DIM){
    float s = 0.f;
    for (int i = 0; i < NNODE; ++i) s += watt[d * NNODE + i];
    wm[d] = s * 0.01f;
  }
}

// ---------------------------------------------------------------- fused attention-score kernel
// One batch's h (25.6KB) is async-copied to LDS per block; per wave one 16-row tile
// of (n,m) pairs, 4 output N-tiles, 64 K-steps of v_wmma_f32_16x16x32_f16.
__global__ __launch_bounds__(128) void kan_score_kernel(
    const float* __restrict__ h, const _Float16* __restrict__ wmat,
    const float* __restrict__ wmean, float* __restrict__ scores){
  __shared__ float shh[NNODE * DIM];          // 25.6 KB
  __shared__ _Float16 stage[4][16][32];       // per-wave A staging
  int tid  = threadIdx.x;
  int wave = tid >> 5;
  int lane = tid & 31;
  int batch = blockIdx.x / 157;
  int tloc  = (blockIdx.x % 157) * 4 + wave;  // 0..627 (625 valid)

  // async copy h[batch] -> LDS (gfx1250 async-to-LDS path, ASYNCcnt tracked)
  {
    unsigned long long gbase = (unsigned long long)(uintptr_t)(h + (size_t)batch * NNODE * DIM);
    unsigned shhOff = (unsigned)(uintptr_t)&shh[0];
    for (int i = tid; i < NNODE * DIM / 4; i += 128){
      unsigned voff = (unsigned)i * 16u;
      unsigned lds  = shhOff + (unsigned)i * 16u;
      asm volatile("global_load_async_to_lds_b128 %0, %1, %2"
                   :: "v"(lds), "v"(voff), "s"(gbase) : "memory");
    }
    asm volatile("s_wait_asynccnt 0x0" ::: "memory");
  }
  __syncthreads();
  if (tloc >= 625) return;

  int rowL = tloc * 16 + (lane & 15);         // 0..9999 within batch
  int n = rowL / 100, m = rowL - n * 100;
  bool hi = lane >= 16;
  _Float16* srow = &stage[wave][lane & 15][0];
  if (lane < 16) { v8h z = {}; *(v8h*)(srow + 24) = z; }   // pad slots stay zero

  v8f c0 = {}, c1 = {}, c2 = {}, c3 = {};
  for (int j = 0; j < DIM; ++j){
    if (lane < 16){
      float x = shh[n * DIM + j] * shh[m * DIM + j];
      stage_feats(x, srow);                   // same-wave LDS ops stay ordered
    }
    v16h af = load_afrag(srow, hi);
    const _Float16* wb = wmat + ((size_t)(j * 4) * 32 + lane) * 16;
    c0 = WMMA_F16(af, *(const v16h*)(wb),        c0);
    c1 = WMMA_F16(af, *(const v16h*)(wb + 512),  c1);
    c2 = WMMA_F16(af, *(const v16h*)(wb + 1024), c2);
    c3 = WMMA_F16(af, *(const v16h*)(wb + 1536), c3);
  }
  int col = lane & 15;
  float w0 = wmean[col], w1 = wmean[col + 16], w2 = wmean[col + 32], w3 = wmean[col + 48];
  float p[8];
#pragma unroll
  for (int v = 0; v < 8; ++v)
    p[v] = tanhf(c0[v]) * w0 + tanhf(c1[v]) * w1 + tanhf(c2[v]) * w2 + tanhf(c3[v]) * w3;
#pragma unroll
  for (int mask = 1; mask < 16; mask <<= 1)
#pragma unroll
    for (int v = 0; v < 8; ++v) p[v] += __shfl_xor(p[v], mask, 32);
  if (col == 0){
    int base = batch * 10000 + tloc * 16 + (hi ? 8 : 0);
#pragma unroll
    for (int v = 0; v < 8; ++v) scores[base + v] = p[v];
  }
}

// ---------------------------------------------------------------- generic WMMA KAN layer on R rows
__global__ __launch_bounds__(128) void kan_rows_kernel(
    const float* __restrict__ X, const _Float16* __restrict__ wmat,
    float* __restrict__ Y, int R, int ntiles, int accum){
  __shared__ _Float16 stage[4][16][32];
  int wave = threadIdx.x >> 5;
  int lane = threadIdx.x & 31;
  int tile = blockIdx.x * 4 + wave;
  if (tile >= ntiles) return;                 // no barriers below -> safe
  int row  = tile * 16 + (lane & 15);
  int rowc = row < R ? row : R - 1;
  const float* xr = X + (size_t)rowc * DIM;
  bool hi = lane >= 16;
  _Float16* srow = &stage[wave][lane & 15][0];
  if (lane < 16) { v8h z = {}; *(v8h*)(srow + 24) = z; }

  v8f c0 = {}, c1 = {}, c2 = {}, c3 = {};
  for (int j = 0; j < DIM; ++j){
    if (lane < 16) stage_feats(xr[j], srow);
    v16h af = load_afrag(srow, hi);
    const _Float16* wb = wmat + ((size_t)(j * 4) * 32 + lane) * 16;
    c0 = WMMA_F16(af, *(const v16h*)(wb),        c0);
    c1 = WMMA_F16(af, *(const v16h*)(wb + 512),  c1);
    c2 = WMMA_F16(af, *(const v16h*)(wb + 1024), c2);
    c3 = WMMA_F16(af, *(const v16h*)(wb + 1536), c3);
  }
  int col = lane & 15;
  int rbase = tile * 16 + (hi ? 8 : 0);
#pragma unroll
  for (int v = 0; v < 8; ++v){
    int rr = rbase + v;
    if (rr < R){
      float* y = Y + (size_t)rr * DIM;
      if (accum){
        y[col]      += c0[v]; y[col + 16] += c1[v];
        y[col + 32] += c2[v]; y[col + 48] += c3[v];
      } else {
        y[col]      = c0[v]; y[col + 16] = c1[v];
        y[col + 32] = c2[v]; y[col + 48] = c3[v];
      }
    }
  }
}

// ---------------------------------------------------------------- softmax + h_att
__global__ __launch_bounds__(64) void att_kernel(const float* __restrict__ scores,
    const float* __restrict__ h, float* __restrict__ hatt){
  __shared__ float sc[NNODE];
  __shared__ float red[2];
  int bn = blockIdx.x; int b = bn / NNODE;
  int tid = threadIdx.x;
  for (int m = tid; m < NNODE; m += 64) sc[m] = scores[bn * NNODE + m];
  __syncthreads();
  if (tid == 0){
    float mx = sc[0];
    for (int m = 1; m < NNODE; ++m) mx = fmaxf(mx, sc[m]);
    red[0] = mx;
  }
  __syncthreads();
  for (int m = tid; m < NNODE; m += 64) sc[m] = expf(sc[m] - red[0]);
  __syncthreads();
  if (tid == 0){
    float z = 0.f;
    for (int m = 0; m < NNODE; ++m) z += sc[m];
    red[1] = 1.f / z;
  }
  __syncthreads();
  float acc = 0.f;
  for (int m = 0; m < NNODE; ++m) acc += sc[m] * h[(b * NNODE + m) * DIM + tid];
  hatt[bn * DIM + tid] = acc * red[1];
}

// ---------------------------------------------------------------- eval BatchNorm
__global__ void bn_kernel(const float* __restrict__ y, const float* __restrict__ gamma,
    const float* __restrict__ beta, float* __restrict__ out){
  int t = blockIdx.x * blockDim.x + threadIdx.x;
  if (t >= 2 * NNODE * DIM) return;
  int d = t & (DIM - 1);
  out[t] = y[t] * (gamma[d] * rsqrtf(1.f + 1e-5f)) + beta[d];
}

// ---------------------------------------------------------------- sigmoid gate + top-50 gather
__global__ __launch_bounds__(128) void gate_topk_kernel(const float* __restrict__ yp,
    const float* __restrict__ hout, float* __restrict__ out){
  __shared__ float s[NNODE];
  int b = blockIdx.x; int tid = threadIdx.x;
  if (tid < NNODE){
    float v = yp[(b * NNODE + tid) * DIM];      // output 0 of the 1-out KAN
    s[tid] = 1.f / (1.f + expf(-v));
  }
  __syncthreads();
  if (tid < NNODE){
    float si = s[tid];
    int rank = 0;
    for (int jj = 0; jj < NNODE; ++jj){
      float sj = s[jj];
      rank += (sj > si) || (sj == si && jj < tid); // lax.top_k tie-break
    }
    if (rank < 50){
      const float* hr  = hout + (b * NNODE + tid) * DIM;
      float* orow = out + (b * 50 + rank) * DIM;
      for (int d = 0; d < DIM; ++d) orow[d] = hr[d] * si;
    }
  }
}

// =================================================================
extern "C" void kernel_launch(void* const* d_in, const int* in_sizes, int n_in,
                              void* d_out, int out_size, void* d_ws, size_t ws_size,
                              hipStream_t stream){
  (void)in_sizes; (void)n_in; (void)out_size; (void)ws_size;
  const float* enc  = (const float*)d_in[0];
  const float* Wt   = (const float*)d_in[1];  const float* bt = (const float*)d_in[2];
  const float* Ws   = (const float*)d_in[3];  const float* bsv = (const float*)d_in[4];
  const float* pe_t = (const float*)d_in[5];  const float* pe_s = (const float*)d_in[6];
  const float* attT[4]  = {(const float*)d_in[7],  (const float*)d_in[8],  (const float*)d_in[9],  (const float*)d_in[10]};
  const float* watt_t   =  (const float*)d_in[11];
  const float* p1T[4]   = {(const float*)d_in[12], (const float*)d_in[13], (const float*)d_in[14], (const float*)d_in[15]};
  const float* p2T[4]   = {(const float*)d_in[16], (const float*)d_in[17], (const float*)d_in[18], (const float*)d_in[19]};
  const float* gamma_t  =  (const float*)d_in[20]; const float* beta_t = (const float*)d_in[21];
  const float* attS[4]  = {(const float*)d_in[22], (const float*)d_in[23], (const float*)d_in[24], (const float*)d_in[25]};
  const float* watt_s   =  (const float*)d_in[26];
  const float* p1S[4]   = {(const float*)d_in[27], (const float*)d_in[28], (const float*)d_in[29], (const float*)d_in[30]};
  const float* p2S[4]   = {(const float*)d_in[31], (const float*)d_in[32], (const float*)d_in[33], (const float*)d_in[34]};
  const float* gamma_s  =  (const float*)d_in[35]; const float* beta_s = (const float*)d_in[36];
  const float* poolT[4] = {(const float*)d_in[37], (const float*)d_in[38], (const float*)d_in[39], (const float*)d_in[40]};
  const float* poolS[4] = {(const float*)d_in[41], (const float*)d_in[42], (const float*)d_in[43], (const float*)d_in[44]};

  float* ws = (float*)d_ws;
  size_t off = 0;
  auto alloc = [&](size_t n) -> float* { float* p = ws + off; off += (n + 7) & ~(size_t)7; return p; };
  float* feat = alloc(2 * 100 * 512);
  float* ht   = alloc(12800); float* hs   = alloc(12800);
  float* wm_t = alloc(64);    float* wm_s = alloc(64);
  float* sc_t = alloc(20000); float* sc_s = alloc(20000);
  float* ha_t = alloc(12800); float* ha_s = alloc(12800);
  float* y_t  = alloc(12800); float* y_s  = alloc(12800);
  float* ho_t = alloc(12800); float* ho_s = alloc(12800);
  float* yp_t = alloc(12800); float* yp_s = alloc(12800);
  const size_t W2N = (size_t)64 * 4 * 32 * 16;          // halves per set
  _Float16* w2 = (_Float16*)(ws + off);
  _Float16* w2_att_t  = w2 + 0 * W2N;  _Float16* w2_p1_t = w2 + 1 * W2N;
  _Float16* w2_p2_t   = w2 + 2 * W2N;  _Float16* w2_pl_t = w2 + 3 * W2N;
  _Float16* w2_att_s  = w2 + 4 * W2N;  _Float16* w2_p1_s = w2 + 5 * W2N;
  _Float16* w2_p2_s   = w2 + 6 * W2N;  _Float16* w2_pl_s = w2 + 7 * W2N;

  feat_kernel<<<(2 * 512 * 100 + 255) / 256, 256, 0, stream>>>(enc, feat);
  proj_kernel<<<200, 64, 0, stream>>>(feat, Wt, bt, pe_t, ht);
  proj_kernel<<<200, 64, 0, stream>>>(feat, Ws, bsv, pe_s, hs);

  const int PG = (64 * 4 * 32 + 255) / 256;
  prep_w2_kernel<<<PG, 256, 0, stream>>>(attT[0], attT[1], attT[2], attT[3], 64, w2_att_t);
  prep_w2_kernel<<<PG, 256, 0, stream>>>(p1T[0],  p1T[1],  p1T[2],  p1T[3],  64, w2_p1_t);
  prep_w2_kernel<<<PG, 256, 0, stream>>>(p2T[0],  p2T[1],  p2T[2],  p2T[3],  64, w2_p2_t);
  prep_w2_kernel<<<PG, 256, 0, stream>>>(poolT[0], poolT[1], poolT[2], poolT[3], 1, w2_pl_t);
  prep_w2_kernel<<<PG, 256, 0, stream>>>(attS[0], attS[1], attS[2], attS[3], 64, w2_att_s);
  prep_w2_kernel<<<PG, 256, 0, stream>>>(p1S[0],  p1S[1],  p1S[2],  p1S[3],  64, w2_p1_s);
  prep_w2_kernel<<<PG, 256, 0, stream>>>(p2S[0],  p2S[1],  p2S[2],  p2S[3],  64, w2_p2_s);
  prep_w2_kernel<<<PG, 256, 0, stream>>>(poolS[0], poolS[1], poolS[2], poolS[3], 1, w2_pl_s);
  wmean_kernel<<<1, 64, 0, stream>>>(watt_t, wm_t);
  wmean_kernel<<<1, 64, 0, stream>>>(watt_s, wm_s);

  // fused KAN attention scores: 2 batches x 157 blocks (4 waves = 4 row-tiles each)
  kan_score_kernel<<<2 * 157, 128, 0, stream>>>(ht, w2_att_t, wm_t, sc_t);
  kan_score_kernel<<<2 * 157, 128, 0, stream>>>(hs, w2_att_s, wm_s, sc_s);

  att_kernel<<<200, 64, 0, stream>>>(sc_t, ht, ha_t);
  att_kernel<<<200, 64, 0, stream>>>(sc_s, hs, ha_s);

  const int RT = 13, RG = (RT + 3) / 4;
  kan_rows_kernel<<<RG, 128, 0, stream>>>(ha_t, w2_p1_t, y_t, 200, RT, 0);
  kan_rows_kernel<<<RG, 128, 0, stream>>>(ht,   w2_p2_t, y_t, 200, RT, 1);
  kan_rows_kernel<<<RG, 128, 0, stream>>>(ha_s, w2_p1_s, y_s, 200, RT, 0);
  kan_rows_kernel<<<RG, 128, 0, stream>>>(hs,   w2_p2_s, y_s, 200, RT, 1);

  bn_kernel<<<(12800 + 255) / 256, 256, 0, stream>>>(y_t, gamma_t, beta_t, ho_t);
  bn_kernel<<<(12800 + 255) / 256, 256, 0, stream>>>(y_s, gamma_s, beta_s, ho_s);

  kan_rows_kernel<<<RG, 128, 0, stream>>>(ho_t, w2_pl_t, yp_t, 200, RT, 0);
  kan_rows_kernel<<<RG, 128, 0, stream>>>(ho_s, w2_pl_s, yp_s, 200, RT, 0);

  float* out_t = (float*)d_out;
  float* out_s = (float*)d_out + 2 * 50 * DIM;
  gate_topk_kernel<<<2, 128, 0, stream>>>(yp_t, ho_t, out_t);
  gate_topk_kernel<<<2, 128, 0, stream>>>(yp_s, ho_s, out_s);
}
